// ComplexResNet_47244640256525
// MI455X (gfx1250) — matrix operand order repaired and checked by reading, not compiled, not run on hardware
//
#include <hip/hip_runtime.h>
#include <hip/hip_bf16.h>
#include <math.h>

typedef __attribute__((ext_vector_type(16))) _Float16 v16h;
typedef __attribute__((ext_vector_type(8)))  _Float16 v8h;
typedef __attribute__((ext_vector_type(8)))  float    v8f;

#define DEV __device__ __forceinline__

static constexpr int BATCH = 262144;
static constexpr int NTILE = BATCH / 16;   // 16 samples per tile (WMMA M dim)

// ---- LDS layout (offsets in halves), all 16B-aligned strides ----
static constexpr int OFF_X  = 0;      // f16 [16][36][8]   input, ch0=xr ch1=xi, pos p = logical l+1
static constexpr int OFF_F1 = 4608;   // f16 [16][37][16]  tanh(r1c1): ch0-7 ar, ch8-15 ai
static constexpr int OFF_F2 = 14080;  // f16 [16][20][16]  pooled stage-1
static constexpr int OFF_F3 = 19200;  // f16 [16][19][8]   tanh(r2c1): ch0-3 ar, ch4-7 ai
static constexpr int OFF_F4 = 21632;  // f16 [16][64]      cr[32] | ci[32]
static constexpr int OFF_L  = 22656;  // f32 [16][40]      lr[20] | li[20]  (1280 halves)
static constexpr int OFF_H1 = 23936;  // f16 [16][32]
static constexpr int OFF_H2 = 24448;  // f16 [16][32]
static constexpr int OFF_BT = 24960;  // f16 [16][32]      B-fragment staging (col-major K)
static constexpr int SM_HALVES = 25472; // 50944 bytes

DEV v8f wmma(v16h a, v16h b, v8f c) {
  return __builtin_amdgcn_wmma_f32_16x16x32_f16(false, a, false, b, (short)0, c, false, false);
}

// Branch-free tanh: clamp + v_exp_f32 + v_rcp_f32 (no EXEC divergence; TRANS ops
// co-execute with the matrix pipe). Accuracy ~1ulp-of-rcp, ample for f16 operands.
DEV float fast_tanh(float x) {
  float xc = fminf(fmaxf(x, -9.0f), 9.0f);
  float e  = __builtin_amdgcn_exp2f(xc * 2.8853900817779268f); // exp(2x)
  return (e - 1.0f) * __builtin_amdgcn_rcpf(e + 1.0f);
}
// Branch-free sigmoid: e = exp(x) (clamped so e stays finite), sig = e/(1+e).
DEV float fast_sig(float x) {
  float xc = fminf(fmaxf(x, -60.0f), 60.0f);
  float e  = __builtin_amdgcn_exp2f(xc * 1.4426950408889634f);
  return e * __builtin_amdgcn_rcpf(e + 1.0f);
}

DEV v8f cinit(float b) {
  v8f c;
#pragma unroll
  for (int i = 0; i < 8; i++) c[i] = b;
  return c;
}
DEV v8f tanh8(v8f x) {
  v8f r;
#pragma unroll
  for (int i = 0; i < 8; i++) r[i] = fast_tanh(x[i]);
  return r;
}
DEV v8f max8(v8f a, v8f b) {
  v8f r;
#pragma unroll
  for (int i = 0; i < 8; i++) r[i] = fmaxf(a[i], b[i]);
  return r;
}
DEV v8f adds8(v8f a, float s) {
  v8f r;
#pragma unroll
  for (int i = 0; i < 8; i++) r[i] = a[i] + s;
  return r;
}

// A-fragment: per-lane two contiguous 16B LDS chunks (documented f16 A 16x32 layout)
DEV v16h afrag(const _Float16* p, int hoff) {
  v8h lo = *(const v8h*)(p + hoff);
  v8h hi = *(const v8h*)(p + hoff + 16);
  return __builtin_shufflevector(lo, hi, 0,1,2,3,4,5,6,7,8,9,10,11,12,13,14,15);
}

// Build a B fragment (32x16 f16): stage col-major-K matrix in LDS, then per-lane load.
// lanes 0-15: col=lane, K 0-7,16-23 ; lanes 16-31: col=lane-16, K 8-15,24-31.
template <typename F>
DEV v16h build_bfrag(_Float16* BT, int lane, F f) {
  for (int idx = lane; idx < 512; idx += 32) {
    int k = idx >> 4, c = idx & 15;
    BT[c * 32 + k] = (_Float16)f(k, c);
  }
  __syncthreads();
  int c = lane & 15, koff = (lane & 16) ? 8 : 0;
  v8h blo = *(const v8h*)(BT + c * 32 + koff);
  v8h bhi = *(const v8h*)(BT + c * 32 + koff + 16);
  v16h r = __builtin_shufflevector(blo, bhi, 0,1,2,3,4,5,6,7,8,9,10,11,12,13,14,15);
  __syncthreads();
  return r;
}

__global__ __launch_bounds__(32) void crnet_kernel(
    const float* __restrict__ x,
    const float* r1c1_wr, const float* r1c1_wi, const float* r1c1_br, const float* r1c1_bi,
    const float* r1c2_wr, const float* r1c2_wi, const float* r1c2_br, const float* r1c2_bi,
    const float* r1sc_wr, const float* r1sc_wi, const float* r1sc_br, const float* r1sc_bi,
    const float* r2c1_wr, const float* r2c1_wi, const float* r2c1_br, const float* r2c1_bi,
    const float* r2c2_wr, const float* r2c2_wi, const float* r2c2_br, const float* r2c2_bi,
    const float* r2sc_wr, const float* r2sc_wi, const float* r2sc_br, const float* r2sc_bi,
    const float* la_wr, const float* la_wi, const float* la_br, const float* la_bi,
    const float* fc1_w, const float* fc1_b, const float* fc2_w, const float* fc2_b,
    const float* fc3_w, const float* fc3_b,
    float* __restrict__ out) {
  __shared__ __align__(128) _Float16 SM[SM_HALVES];
  const int lane  = threadIdx.x;
  const int col   = lane & 15;          // C/B column owned by this lane
  const int rbase = (lane & 16) ? 8 : 0; // C rows r+rbase
  const int hoff  = (lane & 16) ? 8 : 0; // A-fragment half offset
  const int m     = col;                 // A-matrix row (sample in tile)

  // Zero-init whole LDS once (pads must be finite-zero: WMMA propagates NaN)
  {
    v8h z = {};
    for (int i = lane * 8; i < SM_HALVES; i += 32 * 8) *(v8h*)(SM + i) = z;
  }
  __syncthreads();
  _Float16* BT = SM + OFF_BT;

  // ---------------- B fragments (built once, live in VGPRs) ----------------
  // r1c1: K = tap*8 + ch (ch0=xr, ch1=xi); cols 0-7 ar, 8-15 ai
  v16h B_c1 = build_bfrag(BT, lane, [=](int k, int c) -> float {
    int tp = k >> 3, ch = k & 7;
    if (tp >= 3 || ch >= 2) return 0.f;
    int o = c & 7; bool ic = c >= 8;
    float wr = r1c1_wr[o * 3 + tp], wi = r1c1_wi[o * 3 + tp];
    return (ch == 0) ? (ic ? wi : wr) : (ic ? wr : -wi);
  });
  // r1c2: K = tp16*16 + c16 (c<8 ar-ch, c>=8 ai-ch); split taps {0,1} and {2}
  auto c2map = [=](int k, int c, int tb) -> float {
    int tp = tb + (k >> 4); int cc = k & 15;
    if (tp >= 3) return 0.f;
    int i = cc & 7; bool ii = cc >= 8;
    int o = c & 7; bool ic = c >= 8;
    float wr = r1c2_wr[(o * 8 + i) * 3 + tp], wi = r1c2_wi[(o * 8 + i) * 3 + tp];
    return ii ? (ic ? wr : -wi) : (ic ? wi : wr);
  };
  v16h B_c2a = build_bfrag(BT, lane, [=](int k, int c) { return c2map(k, c, 0); });
  v16h B_c2b = build_bfrag(BT, lane, [=](int k, int c) { return c2map(k, c, 2); });
  // r1sc: K0=xr, K1=xi
  v16h B_sc1 = build_bfrag(BT, lane, [=](int k, int c) -> float {
    if (k >= 2) return 0.f;
    int o = c & 7; bool ic = c >= 8;
    float wr = r1sc_wr[o], wi = r1sc_wi[o];
    return (k == 0) ? (ic ? wi : wr) : (ic ? wr : -wi);
  });
  // r2c1 (8->4): cols 0-3 ar, 8-11 ai
  auto m21 = [=](int k, int c, int tb) -> float {
    int tp = tb + (k >> 4); int cc = k & 15;
    if (tp >= 3) return 0.f;
    int i = cc & 7; bool ii = cc >= 8;
    int o = c & 7; if (o >= 4) return 0.f;
    bool ic = c >= 8;
    float wr = r2c1_wr[(o * 8 + i) * 3 + tp], wi = r2c1_wi[(o * 8 + i) * 3 + tp];
    return ii ? (ic ? wr : -wi) : (ic ? wi : wr);
  };
  v16h B_21a = build_bfrag(BT, lane, [=](int k, int c) { return m21(k, c, 0); });
  v16h B_21b = build_bfrag(BT, lane, [=](int k, int c) { return m21(k, c, 2); });
  // r2c2 (4->4): K = tap*8 + c8 (c<4 ar, c>=4 ai) -> single WMMA/position
  v16h B_22 = build_bfrag(BT, lane, [=](int k, int c) -> float {
    int tp = k >> 3, cc = k & 7;
    if (tp >= 3) return 0.f;
    int i = cc & 3; bool ii = cc >= 4;
    int o = c & 7; if (o >= 4) return 0.f;
    bool ic = c >= 8;
    float wr = r2c2_wr[(o * 4 + i) * 3 + tp], wi = r2c2_wi[(o * 4 + i) * 3 + tp];
    return ii ? (ic ? wr : -wi) : (ic ? wi : wr);
  });
  // r2sc (8->4, k1): K = c16 of pooled stage-1
  v16h B_sc2 = build_bfrag(BT, lane, [=](int k, int c) -> float {
    if (k >= 16) return 0.f;
    int i = k & 7; bool ii = k >= 8;
    int o = c & 7; if (o >= 4) return 0.f;
    bool ic = c >= 8;
    float wr = r2sc_wr[o * 8 + i], wi = r2sc_wi[o * 8 + i];
    return ii ? (ic ? wr : -wi) : (ic ? wi : wr);
  });
  // la chunks: c0 -> lr[0..15], c1 -> li[0..15], c2 -> cols0-3 lr[16..19], cols4-7 li[16..19]
  v16h B_la0r = build_bfrag(BT, lane, [=](int k, int c) { return  la_wr[c * 32 + k]; });
  v16h B_la0i = build_bfrag(BT, lane, [=](int k, int c) { return -la_wi[c * 32 + k]; });
  v16h B_la1r = build_bfrag(BT, lane, [=](int k, int c) { return  la_wi[c * 32 + k]; });
  v16h B_la1i = build_bfrag(BT, lane, [=](int k, int c) { return  la_wr[c * 32 + k]; });
  v16h B_la2r = build_bfrag(BT, lane, [=](int k, int c) -> float {
    if (c < 4) return la_wr[(16 + c) * 32 + k];
    if (c < 8) return la_wi[(12 + c) * 32 + k];
    return 0.f;
  });
  v16h B_la2i = build_bfrag(BT, lane, [=](int k, int c) -> float {
    if (c < 4) return -la_wi[(16 + c) * 32 + k];
    if (c < 8) return  la_wr[(12 + c) * 32 + k];
    return 0.f;
  });
  v16h B_f1 = build_bfrag(BT, lane, [=](int k, int c) -> float {
    return (k < 20 && c < 10) ? fc1_w[c * 20 + k] : 0.f; });
  v16h B_f2 = build_bfrag(BT, lane, [=](int k, int c) -> float {
    return (k < 10 && c < 10) ? fc2_w[c * 10 + k] : 0.f; });
  v16h B_f3 = build_bfrag(BT, lane, [=](int k, int c) -> float {
    return (k < 10 && c == 0) ? fc3_w[k] : 0.f; });

  // Per-lane column biases
  const float bias_c1  = (col < 8) ? r1c1_br[col] : r1c1_bi[col - 8];
  const float bias_c2  = (col < 8) ? r1c2_br[col] : r1c2_bi[col - 8];
  const float bias_s1  = (col < 8) ? r1sc_br[col] : r1sc_bi[col - 8];
  const float bias_21  = (col < 4) ? r2c1_br[col] : ((col >= 8 && col < 12) ? r2c1_bi[col - 8] : 0.f);
  const float bias_22  = (col < 4) ? r2c2_br[col] : ((col >= 8 && col < 12) ? r2c2_bi[col - 8] : 0.f);
  const float bias_s2  = (col < 4) ? r2sc_br[col] : ((col >= 8 && col < 12) ? r2sc_bi[col - 8] : 0.f);
  const float bias_la0 = la_br[col];
  const float bias_la1 = la_bi[col];
  const float bias_la2 = (col < 4) ? la_br[16 + col] : ((col < 8) ? la_bi[12 + col] : 0.f);
  const float bias_f1  = (col < 10) ? fc1_b[col] : 0.f;
  const float bias_f2  = (col < 10) ? fc2_b[col] : 0.f;
  const float bias_f3  = (col == 0) ? fc3_b[0] : 0.f;

  const bool w3  = (col < 4) || (col >= 8 && col < 12);
  const int  ch3 = (col < 4) ? col : (col - 4);

  _Float16* Xl  = SM + OFF_X  + m * 36 * 8;   // per-lane sample-m bases for A fragments
  _Float16* F1l = SM + OFF_F1 + m * 37 * 16;
  _Float16* F2l = SM + OFF_F2 + m * 20 * 16;
  _Float16* F3l = SM + OFF_F3 + m * 19 * 8;
  float*    Lf  = (float*)(SM + OFF_L);

  for (int tile = blockIdx.x; tile < NTILE; tile += gridDim.x) {
    __syncthreads();
    // ---- Stage-in: coalesced global f32 -> f16 LDS (pads remain zero) ----
    const float* xt = x + (size_t)tile * 1056;   // 16 samples * 66 floats
    for (int i = lane; i < 1056; i += 32) {
      float v = xt[i];
      int s = i / 66, r = i % 66, part = r / 33, l = r % 33;
      SM[OFF_X + (s * 36 + (l + 1)) * 8 + part] = (_Float16)v;
    }
    if (tile + (int)gridDim.x < NTILE)
      __builtin_prefetch(x + (size_t)(tile + gridDim.x) * 1056, 0, 1);
    __syncthreads();

    // ---- Pass A: r1c1 (1->8 complex, k3) per position, tanh -> F1 ----
    for (int l = 0; l < 33; l++) {
      v8f c = cinit(bias_c1);
      c = wmma(afrag(Xl + l * 8, hoff), B_c1, c);
#pragma unroll
      for (int r = 0; r < 8; r++)
        SM[OFF_F1 + ((r + rbase) * 37 + (l + 1)) * 16 + col] = (_Float16)fast_tanh(c[r]);
    }
    __syncthreads();

    // ---- Pass B: r1c2 + tanh + r1sc (accumulated via C operand) + pool -> F2 ----
    v8f ceven = {};
    for (int l = 0; l < 32; l++) {          // position 32 is dropped by the pool
      v8f c = cinit(bias_c2);
      c = wmma(afrag(F1l + l * 16,       hoff), B_c2a, c);  // taps 0,1
      c = wmma(afrag(F1l + (l + 2) * 16, hoff), B_c2b, c);  // tap 2
      c = adds8(tanh8(c), bias_s1);
      c = wmma(afrag(Xl + (l + 1) * 8, hoff), B_sc1, c);    // shortcut accumulate
      if ((l & 1) == 0) ceven = c;
      else {
        v8f p = max8(ceven, c);
        int pp = (l >> 1) + 1;
#pragma unroll
        for (int r = 0; r < 8; r++)
          SM[OFF_F2 + ((r + rbase) * 20 + pp) * 16 + col] = (_Float16)p[r];
      }
    }
    __syncthreads();

    // ---- Pass C: r2c1 (8->4 complex, k3), tanh -> F3 ----
    for (int l = 0; l < 16; l++) {
      v8f c = cinit(bias_21);
      c = wmma(afrag(F2l + l * 16,       hoff), B_21a, c);
      c = wmma(afrag(F2l + (l + 2) * 16, hoff), B_21b, c);
      if (w3) {
#pragma unroll
        for (int r = 0; r < 8; r++)
          SM[OFF_F3 + ((r + rbase) * 19 + (l + 1)) * 8 + ch3] = (_Float16)fast_tanh(c[r]);
      }
    }
    __syncthreads();

    // ---- Pass D: r2c2 + tanh + r2sc + pool -> F4 (flatten c*8+p) ----
    for (int l = 0; l < 16; l++) {
      v8f c = cinit(bias_22);
      c = wmma(afrag(F3l + l * 8, hoff), B_22, c);
      c = adds8(tanh8(c), bias_s2);
      c = wmma(afrag(F2l + (l + 1) * 16, hoff), B_sc2, c);
      if ((l & 1) == 0) ceven = c;
      else {
        v8f p = max8(ceven, c);
        if (w3) {
          int pos = l >> 1;
          int j = (col < 4) ? (col * 8 + pos) : (32 + (col - 8) * 8 + pos);
#pragma unroll
          for (int r = 0; r < 8; r++)
            SM[OFF_F4 + (r + rbase) * 64 + j] = (_Float16)p[r];
        }
      }
    }
    __syncthreads();

    // ---- Pass E: complex linear (32 -> 20), 3 column-chunks x {cr,ci} ----
    const _Float16* F4l = SM + OFF_F4 + m * 64;
    v16h acr = afrag(F4l,      hoff);
    v16h aci = afrag(F4l + 32, hoff);
    v8f c0 = cinit(bias_la0);
    c0 = wmma(acr, B_la0r, c0); c0 = wmma(aci, B_la0i, c0);
#pragma unroll
    for (int r = 0; r < 8; r++) Lf[(r + rbase) * 40 + col] = c0[r];
    v8f c1 = cinit(bias_la1);
    c1 = wmma(acr, B_la1r, c1); c1 = wmma(aci, B_la1i, c1);
#pragma unroll
    for (int r = 0; r < 8; r++) Lf[(r + rbase) * 40 + 20 + col] = c1[r];
    v8f c2 = cinit(bias_la2);
    c2 = wmma(acr, B_la2r, c2); c2 = wmma(aci, B_la2i, c2);
    if (col < 4) {
#pragma unroll
      for (int r = 0; r < 8; r++) Lf[(r + rbase) * 40 + 16 + col] = c2[r];
    } else if (col < 8) {
#pragma unroll
      for (int r = 0; r < 8; r++) Lf[(r + rbase) * 40 + 32 + col] = c2[r]; // 36+(col-4)
    }
    __syncthreads();

    // ---- fc1: rho = atan(sig(li)/sig(lr)) fused into the A-fragment build ----
    const float* Ls = Lf + m * 40;
    v16h arho;
#pragma unroll
    for (int j = 0; j < 16; j++) {
      int k = (j < 8) ? (hoff + j) : (16 + hoff + (j - 8));
      float v = 0.f;
      if (k < 20) {
        float sr = fast_sig(Ls[k]);
        float si = fast_sig(Ls[20 + k]);
        v = atanf(si * __builtin_amdgcn_rcpf(sr));
      }
      arho[j] = (_Float16)v;
    }
    v8f h = cinit(bias_f1);
    h = wmma(arho, B_f1, h);
#pragma unroll
    for (int r = 0; r < 8; r++)
      SM[OFF_H1 + (r + rbase) * 32 + col] = (_Float16)fast_tanh(h[r]);
    __syncthreads();

    h = cinit(bias_f2);
    h = wmma(afrag(SM + OFF_H1 + m * 32, hoff), B_f2, h);
#pragma unroll
    for (int r = 0; r < 8; r++)
      SM[OFF_H2 + (r + rbase) * 32 + col] = (_Float16)fast_tanh(h[r]);
    __syncthreads();

    h = cinit(bias_f3);
    h = wmma(afrag(SM + OFF_H2 + m * 32, hoff), B_f3, h);
    if (col == 0) {
      float* ob = out + (size_t)tile * 16 + rbase;
#pragma unroll
      for (int r = 0; r < 8; r++) ob[r] = h[r];
    }
  }
}

extern "C" void kernel_launch(void* const* d_in, const int* in_sizes, int n_in,
                              void* d_out, int out_size, void* d_ws, size_t ws_size,
                              hipStream_t stream) {
  (void)in_sizes; (void)n_in; (void)out_size; (void)d_ws; (void)ws_size;
  const float* P[35];
  for (int i = 0; i < 35; i++) P[i] = (const float*)d_in[i];
  crnet_kernel<<<2048, 32, 0, stream>>>(
      P[0],
      P[1], P[2], P[3], P[4],
      P[5], P[6], P[7], P[8],
      P[9], P[10], P[11], P[12],
      P[13], P[14], P[15], P[16],
      P[17], P[18], P[19], P[20],
      P[21], P[22], P[23], P[24],
      P[25], P[26], P[27], P[28],
      P[29], P[30], P[31], P[32],
      P[33], P[34],
      (float*)d_out);
}
